// TSP_Decoder_17033840296103
// MI455X (gfx1250) — compile-verified, matrix operand order checked by reading, not run on hardware
//
#include <hip/hip_runtime.h>
#include <hip/hip_bf16.h>

#define B_    64
#define POMO  200
#define PROB  200
#define EMB   512
#define H_    16
#define D_    32
#define NODES 201   // PROB + 1

typedef unsigned short u16;
typedef unsigned int   u32;

typedef __attribute__((ext_vector_type(16))) __bf16 v16bf;
typedef __attribute__((ext_vector_type(8)))  float  v8f;
typedef __attribute__((ext_vector_type(4)))  float  v4f;
typedef __attribute__((ext_vector_type(4)))  u32    v4u;
typedef __attribute__((ext_vector_type(8)))  u16    v8us;

__device__ __forceinline__ u16 f2bf(float f) {
  u32 u = __float_as_uint(f);
  u32 r = u + 0x7FFFu + ((u >> 16) & 1u);   // round-to-nearest-even
  return (u16)(r >> 16);
}

union FragU  { v16bf v; v4u q[2]; };
union Chunk16 { v4u q; v8us h; };

// Load a 16-half bf16 fragment: this lane's two contiguous 8-half (16B) chunks.
__device__ __forceinline__ v16bf load16(const u16* p0, const u16* p1) {
  FragU f;
  f.q[0] = *(const v4u*)p0;
  f.q[1] = *(const v4u*)p1;
  return f.v;
}

#define WMMA_BF16(a, b, c) \
  __builtin_amdgcn_wmma_f32_16x16x32_bf16(false, (a), false, (b), (short)0, (c), false, false)

// ----- CDNA5 async global->LDS copy (guarded; fallback = vector copy) -------
#if defined(__has_builtin)
# if __has_builtin(__builtin_amdgcn_global_load_async_to_lds_b128)
#  define USE_ASYNC 1
# endif
#endif
#ifndef USE_ASYNC
# define USE_ASYNC 0
#endif

#if USE_ASYNC
typedef int v4i_gcc __attribute__((vector_size(16)));
typedef __attribute__((address_space(1))) v4i_gcc* gptr128;
typedef __attribute__((address_space(3))) v4i_gcc* lptr128;

__device__ __forceinline__ void async_cp16(const u16* g, u16* l) {
  __builtin_amdgcn_global_load_async_to_lds_b128(
      (gptr128)(g), (lptr128)(l), 0, 0);
}
__device__ __forceinline__ void wait_async0() {
# if __has_builtin(__builtin_amdgcn_s_wait_asynccnt)
  __builtin_amdgcn_s_wait_asynccnt(0);
# else
  asm volatile("s_wait_asynccnt 0x0" ::: "memory");
# endif
}
#else
__device__ __forceinline__ void async_cp16(const u16* g, u16* l) {
  *(v4u*)l = *(const v4u*)g;
}
__device__ __forceinline__ void wait_async0() {}
#endif

// ---------------------------------------------------------------------------
// Kernel 0: f32 -> bf16, 8 elements per thread iteration (packed b128 stores).
// ---------------------------------------------------------------------------
__global__ void cvt_bf16x8_kernel(const float* __restrict__ in, u16* __restrict__ out, int n8) {
  int i = blockIdx.x * blockDim.x + threadIdx.x;
  int stride = gridDim.x * blockDim.x;
  for (; i < n8; i += stride) {
    v4f a = *(const v4f*)(in + (size_t)i * 8);
    v4f b = *(const v4f*)(in + (size_t)i * 8 + 4);
    v4u o;
    o.x = (u32)f2bf(a.x) | ((u32)f2bf(a.y) << 16);
    o.y = (u32)f2bf(a.z) | ((u32)f2bf(a.w) << 16);
    o.z = (u32)f2bf(b.x) | ((u32)f2bf(b.y) << 16);
    o.w = (u32)f2bf(b.z) | ((u32)f2bf(b.w) << 16);
    *(v4u*)(out + (size_t)i * 8) = o;
  }
}

// ---------------------------------------------------------------------------
// Kernel 0b: weight transpose+convert: Wt[n][k] = bf16(W[k][n]), 512x512.
// ---------------------------------------------------------------------------
__launch_bounds__(256)
__global__ void wtrans_kernel(const float* __restrict__ W, u16* __restrict__ Wt) {
  __shared__ u16 tile[32][33];
  const int k0 = blockIdx.x * 32, n0 = blockIdx.y * 32;
  const int c = threadIdx.x & 31, r0 = threadIdx.x >> 5;
  for (int rr = r0; rr < 32; rr += 8)
    tile[c][rr] = f2bf(W[(size_t)(k0 + rr) * EMB + n0 + c]);
  __syncthreads();
  for (int rr = r0; rr < 32; rr += 8)
    Wt[(size_t)(n0 + rr) * EMB + k0 + c] = tile[rr][c];
}

// ---------------------------------------------------------------------------
// GEMM: out = A1@W1t^T (+A2@W2t^T) (+bias), all operands bf16 K-contiguous.
// No LDS, no barriers. Wave computes a 32x64 tile: per K-step 2 A frags +
// 4 shared B frags -> 8 WMMAs (1.5 b128 loads per WMMA). Fragments are
// double-buffered so loads of step ks+1 overlap the WMMAs of step ks.
// M must be a multiple of 32 (true for all call sites: 12864, 12800).
// ---------------------------------------------------------------------------
__launch_bounds__(256, 1)
__global__ void gemm_nt_bf16_kernel(const u16* __restrict__ A1, const u16* __restrict__ Wt1,
                                    const u16* __restrict__ A2, const u16* __restrict__ Wt2,
                                    const float* __restrict__ bias,
                                    u16* __restrict__ out, int M)
{
  const int lane = threadIdx.x & 31, wave = threadIdx.x >> 5;
  const int lr = lane & 15, lh = lane >> 4;
  const int row0 = blockIdx.y * 256 + wave * 32;
  const int col0 = blockIdx.x * 64;
  if (row0 >= M) return;                            // wave-uniform

  v8f acc[2][4] = {{{}, {}, {}, {}}, {{}, {}, {}, {}}};
  const int npass = (A2 != nullptr) ? 2 : 1;
  for (int pass = 0; pass < npass; ++pass) {
    const u16* A  = pass ? A2 : A1;
    const u16* Wt = pass ? Wt2 : Wt1;
    const u16* a0row = A + (size_t)(row0 + lr) * EMB + lh * 8;
    const u16* a1row = a0row + 16 * EMB;
    const u16* brow  = Wt + (size_t)(col0 + lr) * EMB + lh * 8;

    v16bf a0c = load16(a0row, a0row + 16);
    v16bf a1c = load16(a1row, a1row + 16);
    v16bf bc[4];
#pragma unroll
    for (int j = 0; j < 4; ++j) {
      const u16* bp = brow + (size_t)j * 16 * EMB;
      bc[j] = load16(bp, bp + 16);
    }
#pragma unroll
    for (int ks = 0; ks < 16; ++ks) {
      v16bf a0n, a1n, bn[4];
      if (ks < 15) {                                // prefetch next K-step
        const u16* pa0 = a0row + (ks + 1) * 32;
        const u16* pa1 = a1row + (ks + 1) * 32;
        a0n = load16(pa0, pa0 + 16);
        a1n = load16(pa1, pa1 + 16);
#pragma unroll
        for (int j = 0; j < 4; ++j) {
          const u16* bp = brow + (size_t)j * 16 * EMB + (ks + 1) * 32;
          bn[j] = load16(bp, bp + 16);
        }
      }
#pragma unroll
      for (int j = 0; j < 4; ++j) {
        acc[0][j] = WMMA_BF16(a0c, bc[j], acc[0][j]);
        acc[1][j] = WMMA_BF16(a1c, bc[j], acc[1][j]);
      }
      if (ks < 15) {
        a0c = a0n; a1c = a1n;
#pragma unroll
        for (int j = 0; j < 4; ++j) bc[j] = bn[j];
      }
    }
  }
#pragma unroll
  for (int g = 0; g < 2; ++g) {
#pragma unroll
    for (int j = 0; j < 4; ++j) {
      int col = col0 + j * 16 + lr;
      float bj = bias ? bias[col] : 0.f;
#pragma unroll
      for (int r = 0; r < 8; ++r) {
        int row = row0 + g * 16 + lh * 8 + r;
        out[(size_t)row * EMB + col] = f2bf(acc[g][j][r] + bj);
      }
    }
  }
}

// ---------------------------------------------------------------------------
// Attention per (b,h): K/V^T staged to LDS (async b128 where available);
// per-wave 16-row f32 score slab -> softmax -> P@V. ~203KB dynamic LDS.
// ---------------------------------------------------------------------------
__launch_bounds__(256, 1)
__global__ void attn_kernel(const u16* __restrict__ qg,
                            const u16* __restrict__ kg,
                            const u16* __restrict__ vg,
                            const float* __restrict__ ninf,
                            u16* __restrict__ outc)
{
  extern __shared__ char smem[];
  u16*   kslab    = (u16*)smem;                      // 208 x 32 bf16
  u16*   vtslab   = kslab + 208 * 32;                // 32 x 224 bf16 (V^T)
  float* scoreAll = (float*)(vtslab + 32 * 224);     // 8 * 16 x 224 f32
  u16*   pAll     = (u16*)(scoreAll + 8 * 16 * 224); // 8 * 16 x 224 bf16
  u16*   qAll     = pAll + 8 * 16 * 224;             // 8 * 16 x 32 bf16

  const int h = blockIdx.x, b = blockIdx.y;
  const int tid = threadIdx.x, lane = tid & 31, wave = tid >> 5;
  const int lr = lane & 15, lh = lane >> 4;

  // stage K: 208 rows * 4 chunks of 16B
  for (int i = tid; i < 208 * 4; i += 256) {
    int n = i >> 2, c = i & 3;
    if (n < NODES)
      async_cp16(&kg[((size_t)(b * NODES + n)) * EMB + h * D_ + c * 8], &kslab[n * 32 + c * 8]);
    else
      *(v4u*)&kslab[n * 32 + c * 8] = (v4u)0u;
  }
  // stage V transposed: read 16B chunk of a row, scatter 8 halfs
  for (int i = tid; i < 224 * 4; i += 256) {
    int n = i >> 2, c = i & 3;
    if (n < NODES) {
      Chunk16 x;
      x.q = *(const v4u*)&vg[((size_t)(b * NODES + n)) * EMB + h * D_ + c * 8];
#pragma unroll
      for (int j = 0; j < 8; ++j) vtslab[(c * 8 + j) * 224 + n] = x.h[j];
    } else {
#pragma unroll
      for (int j = 0; j < 8; ++j) vtslab[(c * 8 + j) * 224 + n] = 0;
    }
  }
  wait_async0();
  __syncthreads();

  float* score = scoreAll + wave * (16 * 224);
  u16*   pslab = pAll     + wave * (16 * 224);
  u16*   qslab = qAll     + wave * (16 * 32);
  const float scale = 0.17677669529663687f;          // 1/sqrt(D)

  for (int t = wave; t < 13; t += 8) {               // wave-uniform tile loop
    int qrow0 = t * 16;
    for (int i = lane; i < 16 * 4; i += 32) {        // stage q tile (16 rows * 4 chunks)
      int r = i >> 2, c = i & 3;
      int row = qrow0 + r;
      if (row < POMO)
        async_cp16(&qg[((size_t)(b * POMO + row)) * EMB + h * D_ + c * 8], &qslab[r * 32 + c * 8]);
      else
        *(v4u*)&qslab[r * 32 + c * 8] = (v4u)0u;
    }
    wait_async0();
    asm volatile("s_wait_dscnt 0x0" ::: "memory");

    const u16* ap = &qslab[lr * 32 + lh * 8];
    v16bf a = load16(ap, ap + 16);

    for (int kt = 0; kt < 13; ++kt) {                // score = q @ k^T
      const u16* bp = &kslab[(kt * 16 + lr) * 32 + lh * 8];
      v16bf bfr = load16(bp, bp + 16);
      v8f c = {};
      c = WMMA_BF16(a, bfr, c);
      int col = kt * 16 + lr;
#pragma unroll
      for (int r = 0; r < 8; ++r) {
        int m = lh * 8 + r;
        int qrow = qrow0 + m; if (qrow > POMO - 1) qrow = POMO - 1;
        float mask = (col < PROB) ? ninf[((size_t)(b * POMO + qrow)) * PROB + col] : 0.f;
        score[m * 224 + col] = c[r] * scale + mask;
      }
    }
    asm volatile("s_wait_dscnt 0x0" ::: "memory");

    {                                                // row softmax (lane pair / row)
      int row = lr, half = lh;
      float mx = -3.0e38f;
      for (int c = half; c < NODES; c += 2) mx = fmaxf(mx, score[row * 224 + c]);
      mx = fmaxf(mx, __shfl_xor(mx, 16, 32));
      float sum = 0.f;
      for (int c = half; c < NODES; c += 2) sum += __expf(score[row * 224 + c] - mx);
      sum += __shfl_xor(sum, 16, 32);
      float inv = 1.0f / sum;
      for (int c = half; c < 224; c += 2) {
        float p = (c < NODES) ? __expf(score[row * 224 + c] - mx) * inv : 0.f;
        pslab[row * 224 + c] = f2bf(p);
      }
    }
    asm volatile("s_wait_dscnt 0x0" ::: "memory");

    v8f o0 = {}, o1 = {};                            // out = P @ V
    for (int s = 0; s < 7; ++s) {
      const u16* pp  = &pslab[lr * 224 + s * 32 + lh * 8];
      v16bf pa = load16(pp, pp + 16);
      const u16* v0p = &vtslab[lr * 224 + s * 32 + lh * 8];
      const u16* v1p = &vtslab[(16 + lr) * 224 + s * 32 + lh * 8];
      v16bf vb0 = load16(v0p, v0p + 16);
      v16bf vb1 = load16(v1p, v1p + 16);
      o0 = WMMA_BF16(pa, vb0, o0);
      o1 = WMMA_BF16(pa, vb1, o1);
    }
#pragma unroll
    for (int r = 0; r < 8; ++r) {
      int row = qrow0 + lh * 8 + r;
      if (row < POMO) {
        size_t base = ((size_t)(b * POMO + row)) * EMB + h * D_;
        outc[base + lr]      = f2bf(o0[r]);
        outc[base + 16 + lr] = f2bf(o1[r]);
      }
    }
  }
}

// ---------------------------------------------------------------------------
// score2 = mh @ nodes[:, :-1]^T / sqrt(EMB); 10*tanh + mask; row softmax.
// One wave per (b, 16-row n-tile); A fragments hoisted across full K=512;
// B fragments double-buffered against the WMMA chain.
// ---------------------------------------------------------------------------
__launch_bounds__(32, 1)
__global__ void score2_softmax_kernel(const u16* __restrict__ mh,
                                      const u16* __restrict__ nodes_bf,
                                      const float* __restrict__ ninf,
                                      float* __restrict__ out)
{
  __shared__ float slab[16 * 208];
  const int t = blockIdx.x;
  const int b = blockIdx.y;
  const int lane = threadIdx.x & 31;
  const int lr = lane & 15, lh = lane >> 4;
  const int n0 = t * 16;

  v16bf afr[16];
  {
    int row = n0 + lr; if (row > POMO - 1) row = POMO - 1;
    const u16* base = &mh[((size_t)(b * POMO + row)) * EMB];
#pragma unroll
    for (int ks = 0; ks < 16; ++ks) {
      const u16* p = base + ks * 32 + lh * 8;
      afr[ks] = load16(p, p + 16);
    }
  }

  const float invs = (float)(1.0 / 22.627416997969522);
  for (int pt = 0; pt < 13; ++pt) {
    int pcol = pt * 16 + lr;
    int prow = pcol; if (prow > PROB - 1) prow = PROB - 1;
    const u16* nb = &nodes_bf[((size_t)(b * NODES + prow)) * EMB] + lh * 8;
    v8f acc = {};
    v16bf bcur = load16(nb, nb + 16);
#pragma unroll
    for (int ks = 0; ks < 16; ++ks) {
      v16bf bnxt;
      if (ks < 15) {
        const u16* p = nb + (ks + 1) * 32;
        bnxt = load16(p, p + 16);
      }
      acc = WMMA_BF16(afr[ks], bcur, acc);
      if (ks < 15) bcur = bnxt;
    }
#pragma unroll
    for (int r = 0; r < 8; ++r) {
      int m = lh * 8 + r;
      int qn = n0 + m; if (qn > POMO - 1) qn = POMO - 1;
      float s = 10.0f * tanhf(acc[r] * invs);
      if (pcol < PROB) s += ninf[((size_t)(b * POMO + qn)) * PROB + pcol];
      slab[m * 208 + pt * 16 + lr] = s;
    }
  }
  asm volatile("s_wait_dscnt 0x0" ::: "memory");

  {
    int row = lr, half = lh;
    int qn = n0 + row;
    float mx = -3.0e38f;
    for (int c = half; c < PROB; c += 2) mx = fmaxf(mx, slab[row * 208 + c]);
    mx = fmaxf(mx, __shfl_xor(mx, 16, 32));
    float sum = 0.f;
    for (int c = half; c < PROB; c += 2) sum += __expf(slab[row * 208 + c] - mx);
    sum += __shfl_xor(sum, 16, 32);
    float inv = 1.0f / sum;
    if (qn < POMO) {
      float* orow = out + ((size_t)(b * POMO + qn)) * PROB;
      for (int c = half; c < PROB; c += 2)
        orow[c] = __expf(slab[row * 208 + c] - mx) * inv;
    }
  }
}

// ---------------------------------------------------------------------------
extern "C" void kernel_launch(void* const* d_in, const int* in_sizes, int n_in,
                              void* d_out, int out_size, void* d_ws, size_t ws_size,
                              hipStream_t stream)
{
  (void)in_sizes; (void)n_in; (void)out_size; (void)ws_size;
  const float* enc_nodes = (const float*)d_in[0];
  const float* enc_q1    = (const float*)d_in[1];
  const float* enc_last  = (const float*)d_in[2];
  const float* ninf      = (const float*)d_in[3];
  const float* Wq_first  = (const float*)d_in[4];
  const float* Wq_last   = (const float*)d_in[5];
  const float* Wk        = (const float*)d_in[6];
  const float* Wv        = (const float*)d_in[7];
  const float* Wc        = (const float*)d_in[8];
  const float* bc        = (const float*)d_in[9];
  float* outp = (float*)d_out;

  char* ws = (char*)d_ws;
  size_t off = 0;
  const size_t MN = (size_t)B_ * NODES;   // 12864
  const size_t MP = (size_t)B_ * POMO;    // 12800
  u16* nodes_bf = (u16*)(ws + off); off += MN * EMB * 2;
  u16* q1_bf    = (u16*)(ws + off); off += MP * EMB * 2;
  u16* last_bf  = (u16*)(ws + off); off += MP * EMB * 2;
  u16* k_bf     = (u16*)(ws + off); off += MN * EMB * 2;
  u16* v_bf     = (u16*)(ws + off); off += MN * EMB * 2;
  u16* q_bf     = (u16*)(ws + off); off += MP * EMB * 2;
  u16* wt_qf    = (u16*)(ws + off); off += (size_t)EMB * EMB * 2;
  u16* wt_ql    = (u16*)(ws + off); off += (size_t)EMB * EMB * 2;
  u16* wt_k     = (u16*)(ws + off); off += (size_t)EMB * EMB * 2;
  u16* wt_v     = (u16*)(ws + off); off += (size_t)EMB * EMB * 2;
  u16* wt_c     = (u16*)(ws + off); off += (size_t)EMB * EMB * 2;
  u16* outc_bf  = q1_bf;    // dead after q projection -> reuse
  u16* mh_bf    = last_bf;  // dead after q projection -> reuse

  // --- bf16 conversions / weight transposes ---
  cvt_bf16x8_kernel<<<512, 256, 0, stream>>>(enc_nodes, nodes_bf, (int)(MN * EMB / 8));
  cvt_bf16x8_kernel<<<512, 256, 0, stream>>>(enc_q1,    q1_bf,    (int)(MP * EMB / 8));
  cvt_bf16x8_kernel<<<512, 256, 0, stream>>>(enc_last,  last_bf,  (int)(MP * EMB / 8));
  {
    dim3 g(16, 16);
    wtrans_kernel<<<g, 256, 0, stream>>>(Wq_first, wt_qf);
    wtrans_kernel<<<g, 256, 0, stream>>>(Wq_last,  wt_ql);
    wtrans_kernel<<<g, 256, 0, stream>>>(Wk,       wt_k);
    wtrans_kernel<<<g, 256, 0, stream>>>(Wv,       wt_v);
    wtrans_kernel<<<g, 256, 0, stream>>>(Wc,       wt_c);
  }
  // --- K/V/Q projections ---
  {
    dim3 g(EMB / 64, (unsigned)((MN + 255) / 256));
    gemm_nt_bf16_kernel<<<g, 256, 0, stream>>>(nodes_bf, wt_k, nullptr, nullptr, nullptr, k_bf, (int)MN);
    gemm_nt_bf16_kernel<<<g, 256, 0, stream>>>(nodes_bf, wt_v, nullptr, nullptr, nullptr, v_bf, (int)MN);
  }
  {
    dim3 g(EMB / 64, (unsigned)((MP + 255) / 256));
    gemm_nt_bf16_kernel<<<g, 256, 0, stream>>>(q1_bf, wt_qf, last_bf, wt_ql, nullptr, q_bf, (int)MP);
  }
  // --- attention ---
  {
    dim3 g(H_, B_);
    size_t smem = (size_t)(208 * 32 + 32 * 224) * 2
                + (size_t)8 * 16 * 224 * 4
                + (size_t)8 * 16 * 224 * 2
                + (size_t)8 * 16 * 32 * 2;
    attn_kernel<<<g, 256, smem, stream>>>(q_bf, k_bf, v_bf, ninf, outc_bf);
  }
  // --- output projection ---
  {
    dim3 g(EMB / 64, (unsigned)((MP + 255) / 256));
    gemm_nt_bf16_kernel<<<g, 256, 0, stream>>>(outc_bf, wt_c, nullptr, nullptr, bc, mh_bf, (int)MP);
  }
  // --- score2 + clip-tanh + softmax ---
  {
    dim3 g(13, B_);
    score2_softmax_kernel<<<g, 32, 0, stream>>>(mh_bf, nodes_bf, ninf, outp);
  }
}
